// DeepseekV4Attention_74783970558182
// MI455X (gfx1250) — matrix operand order repaired and checked by reading, not compiled, not run on hardware
//
#include <hip/hip_runtime.h>

#define S_LEN   2048
#define D_MODEL 4096
#define NH      32
#define DHD     128
#define QLRN    1536
#define OLRN    512
#define NG      4

typedef __attribute__((ext_vector_type(16))) __bf16 bf16x16;
typedef __attribute__((ext_vector_type(8)))  __bf16 bf16x8;
typedef __attribute__((ext_vector_type(8)))  float  f32x8;

// ---------------- WMMA fragment helpers (contiguous k order, same for A and B) ----------------
// A operand 16x32 (MxK): lane m=lane&15; K = 16*(lane>>4) .. +15 contiguous.
// p points at the row's K-chunk start.
static __device__ __forceinline__ bf16x16 frag_a32(const __bf16* p, int lh) {
  bf16x8 lo = *(const bf16x8*)(p + 16 * lh);
  bf16x8 hi = *(const bf16x8*)(p + 16 * lh + 8);
  return __builtin_shufflevector(lo, hi, 0,1,2,3,4,5,6,7,8,9,10,11,12,13,14,15);
}
// B operand 32x16 (KxN): lane n=lane&15 holds column n; K = 16*(lane>>4) .. +15 contiguous.
// p points at the start of the 32-K chunk of the "column row" (weights stored [N,K]).
static __device__ __forceinline__ bf16x16 frag_b32(const __bf16* p, int lh) {
  bf16x8 lo = *(const bf16x8*)(p + 16 * lh);
  bf16x8 hi = *(const bf16x8*)(p + 16 * lh + 8);
  return __builtin_shufflevector(lo, hi, 0,1,2,3,4,5,6,7,8,9,10,11,12,13,14,15);
}
static __device__ __forceinline__ f32x8 wmma_bf16(bf16x16 a, bf16x16 b, f32x8 c) {
  return __builtin_amdgcn_wmma_f32_16x16x32_bf16(false, a, false, b, (short)0, c, false, false);
}

// ---------------- f32 -> bf16 convert ----------------
__global__ void cvt_f32_bf16(const float* __restrict__ s, __bf16* __restrict__ d, long long n) {
  long long i = (long long)blockIdx.x * blockDim.x + threadIdx.x;
  long long st = (long long)gridDim.x * blockDim.x;
  for (; i < n; i += st) d[i] = (__bf16)s[i];
}

// ---------------- generic GEMM: C[M,N] = A[M,K] * B[N,K]^T, bf16 in, f32 out ----------------
// block = 256 threads (8 waves), block tile 128x128, wave tile 32x64.
__global__ __launch_bounds__(256) void gemm_bt(
    const __bf16* __restrict__ A, const __bf16* __restrict__ B, float* __restrict__ C,
    int M, int N, int K, int lda, int ldb, int ldc,
    long long aBatch, long long bBatch, long long cBatch) {
  A += (long long)blockIdx.z * aBatch;
  B += (long long)blockIdx.z * bBatch;
  C += (long long)blockIdx.z * cBatch;
  const int lane = threadIdx.x & 31, wave = threadIdx.x >> 5;
  const int lm = lane & 15, lh = lane >> 4;
  const int m0 = blockIdx.x * 128 + (wave & 3) * 32;
  const int n0 = blockIdx.y * 128 + (wave >> 2) * 64;

  f32x8 acc[2][4];
#pragma unroll
  for (int mt = 0; mt < 2; mt++)
#pragma unroll
    for (int nt = 0; nt < 4; nt++) acc[mt][nt] = (f32x8)0.0f;

  const __bf16* arow[2];
#pragma unroll
  for (int mt = 0; mt < 2; mt++) arow[mt] = A + (long long)(m0 + 16 * mt + lm) * lda;
  const __bf16* brow[4];
#pragma unroll
  for (int nt = 0; nt < 4; nt++) brow[nt] = B + (long long)(n0 + 16 * nt + lm) * ldb;

  for (int k0 = 0; k0 < K; k0 += 32) {
    bf16x16 af[2];
#pragma unroll
    for (int mt = 0; mt < 2; mt++) af[mt] = frag_a32(arow[mt] + k0, lh);
    bf16x16 bfg[4];
#pragma unroll
    for (int nt = 0; nt < 4; nt++) bfg[nt] = frag_b32(brow[nt] + k0, lh);
#pragma unroll
    for (int mt = 0; mt < 2; mt++)
#pragma unroll
      for (int nt = 0; nt < 4; nt++) acc[mt][nt] = wmma_bf16(af[mt], bfg[nt], acc[mt][nt]);
  }
#pragma unroll
  for (int mt = 0; mt < 2; mt++)
#pragma unroll
    for (int nt = 0; nt < 4; nt++)
#pragma unroll
      for (int r = 0; r < 8; r++)
        C[(long long)(m0 + 16 * mt + 8 * lh + r) * ldc + (n0 + 16 * nt + lm)] = acc[mt][nt][r];
}

// ---------------- RMS norm (row-wise) f32 -> bf16 ----------------
__global__ __launch_bounds__(256) void rmsnorm_bf16(const float* __restrict__ src,
                                                    const float* __restrict__ g,
                                                    __bf16* __restrict__ dst, int C) {
  const long long row = blockIdx.x;
  const float* x = src + row * C;
  float ss = 0.f;
  for (int c = threadIdx.x; c < C; c += 256) { float v = x[c]; ss += v * v; }
#pragma unroll
  for (int m = 16; m; m >>= 1) ss += __shfl_xor(ss, m, 32);
  __shared__ float red[8];
  if ((threadIdx.x & 31) == 0) red[threadIdx.x >> 5] = ss;
  __syncthreads();
  float tot = 0.f;
#pragma unroll
  for (int i = 0; i < 8; i++) tot += red[i];
  const float scale = rsqrtf(tot / (float)C + 1e-6f);
  __bf16* y = dst + row * C;
  for (int c = threadIdx.x; c < C; c += 256) y[c] = (__bf16)(x[c] * scale * g[c]);
}

// ---------------- kv epilogue: rmsnorm, write V^T, write roped K ----------------
__global__ __launch_bounds__(128) void kv_epilogue(const float* __restrict__ kvf,
                                                   const float* __restrict__ g,
                                                   __bf16* __restrict__ vT,
                                                   __bf16* __restrict__ kbf) {
  const int s = blockIdx.x, c = threadIdx.x;  // 128 threads
  __shared__ float row[DHD];
  __shared__ float part[4];
  float v = kvf[(long long)s * DHD + c];
  float sq = v * v;
#pragma unroll
  for (int m = 16; m; m >>= 1) sq += __shfl_xor(sq, m, 32);
  if ((c & 31) == 0) part[c >> 5] = sq;
  __syncthreads();
  const float tot = part[0] + part[1] + part[2] + part[3];
  const float scale = rsqrtf(tot / (float)DHD + 1e-6f);
  const float val = v * scale * g[c];
  row[c] = val;
  __syncthreads();
  vT[(long long)c * S_LEN + s] = (__bf16)val;   // V transposed: [d][t]
  float kout = val;
  if (c >= 64) {
    const int i = (c - 64) >> 1;
    const float freq = __powf(10000.f, -(float)(2 * i) / 64.f);
    const float ang = (float)s * freq;
    const float cs = __cosf(ang), sn = __sinf(ang);
    kout = ((c & 1) == 0) ? (row[c] * cs - row[c + 1] * sn)
                          : (row[c - 1] * sn + row[c] * cs);
  }
  kbf[(long long)s * DHD + c] = (__bf16)kout;
}

// ---------------- q epilogue: rope last 64 dims per head, fold in 1/sqrt(DH) ----------------
__global__ __launch_bounds__(64) void q_rope(const float* __restrict__ qf, __bf16* __restrict__ qb) {
  const int s = blockIdx.x, h = blockIdx.y, t = threadIdx.x;  // 64 threads, 2 elems each
  const float scale = 0.08838834764831845f;                    // DH^-0.5
  const float* x = qf + ((long long)s * NH + h) * DHD;
  __bf16* y = qb + ((long long)s * NH + h) * DHD;
  const int c0 = 2 * t;
  float a = x[c0], b = x[c0 + 1];
  float r0 = a, r1 = b;
  if (c0 >= 64) {
    const int i = (c0 - 64) >> 1;
    const float freq = __powf(10000.f, -(float)(2 * i) / 64.f);
    const float ang = (float)s * freq;
    const float cs = __cosf(ang), sn = __sinf(ang);
    r0 = a * cs - b * sn;
    r1 = a * sn + b * cs;
  }
  y[c0] = (__bf16)(r0 * scale);
  y[c0 + 1] = (__bf16)(r1 * scale);
}

// ---------------- flash attention with sink, causal, WMMA ----------------
// grid = (S/128, H), block = 256 (8 waves). Each wave: 16 query rows of one head.
__global__ __launch_bounds__(256) void attn_fwd(const __bf16* __restrict__ q,
                                                const __bf16* __restrict__ k,
                                                const __bf16* __restrict__ vT,
                                                const float* __restrict__ sink,
                                                __bf16* __restrict__ out) {
  __shared__ __bf16 lds_p[8][16 * 64];
  const int lane = threadIdx.x & 31, w = threadIdx.x >> 5;
  const int lm = lane & 15, lh = lane >> 4;
  const int h = blockIdx.y;
  const int qbase = blockIdx.x * 128 + w * 16;

  bf16x16 qf[4];
  {
    const __bf16* qrow = q + (long long)(qbase + lm) * (NH * DHD) + h * DHD;
#pragma unroll
    for (int kk = 0; kk < 4; kk++) qf[kk] = frag_a32(qrow + 32 * kk, lh);
  }
  f32x8 oacc[8];
#pragma unroll
  for (int dt = 0; dt < 8; dt++) oacc[dt] = (f32x8)0.0f;
  float mi[8], li[8];
  const float snk = sink[h];
#pragma unroll
  for (int r = 0; r < 8; r++) { mi[r] = snk; li[r] = 1.0f; }  // sink = initial softmax state

  for (int t0 = 0; t0 <= qbase + 15; t0 += 64) {
    // scores: 4 tiles of 16x16 over K=128
    f32x8 sc[4];
#pragma unroll
    for (int j = 0; j < 4; j++) {
      f32x8 s = (f32x8)0.0f;
      const __bf16* krow = k + (long long)(t0 + 16 * j + lm) * DHD;
#pragma unroll
      for (int kk = 0; kk < 4; kk++) s = wmma_bf16(qf[kk], frag_b32(krow + 32 * kk, lh), s);
      const int tcol = t0 + 16 * j + lm;
#pragma unroll
      for (int r = 0; r < 8; r++) {
        const int srow = qbase + 8 * lh + r;
        if (tcol > srow) s[r] = -3.0e38f;  // causal mask (WIN >= S)
      }
      sc[j] = s;
    }
    // online softmax update
    float corr[8];
#pragma unroll
    for (int r = 0; r < 8; r++) {
      float v = fmaxf(fmaxf(sc[0][r], sc[1][r]), fmaxf(sc[2][r], sc[3][r]));
#pragma unroll
      for (int m = 8; m; m >>= 1) v = fmaxf(v, __shfl_xor(v, m, 32));
      const float mnew = fmaxf(mi[r], v);
      corr[r] = __expf(mi[r] - mnew);
      li[r] *= corr[r];
      mi[r] = mnew;
    }
#pragma unroll
    for (int dt = 0; dt < 8; dt++)
#pragma unroll
      for (int r = 0; r < 8; r++) oacc[dt][r] *= corr[r];
    float psum[8];
#pragma unroll
    for (int r = 0; r < 8; r++) psum[r] = 0.f;
#pragma unroll
    for (int j = 0; j < 4; j++)
#pragma unroll
      for (int r = 0; r < 8; r++) {
        const float p = __expf(sc[j][r] - mi[r]);
        psum[r] += p;
        lds_p[w][(8 * lh + r) * 64 + 16 * j + lm] = (__bf16)p;  // C-layout -> row-major in LDS
      }
#pragma unroll
    for (int r = 0; r < 8; r++) {
      float v = psum[r];
#pragma unroll
      for (int m = 8; m; m >>= 1) v += __shfl_xor(v, m, 32);
      li[r] += v;
    }
    // wave-local LDS relayout barrier (no block sync: waves have different trip counts)
    asm volatile("s_wait_dscnt 0" ::: "memory");
    bf16x16 pf[2];
#pragma unroll
    for (int kk = 0; kk < 2; kk++) pf[kk] = frag_a32(&lds_p[w][lm * 64 + 32 * kk], lh);
    // O += P @ V  (V^T gives contiguous B-operand loads)
#pragma unroll
    for (int dt = 0; dt < 8; dt++) {
#pragma unroll
      for (int kk = 0; kk < 2; kk++) {
        const __bf16* vrow = vT + (long long)(16 * dt + lm) * S_LEN + t0 + 32 * kk;
        oacc[dt] = wmma_bf16(pf[kk], frag_b32(vrow, lh), oacc[dt]);
      }
    }
  }
#pragma unroll
  for (int dt = 0; dt < 8; dt++)
#pragma unroll
    for (int r = 0; r < 8; r++) {
      const float o = oacc[dt][r] / li[r];
      out[(long long)(qbase + 8 * lh + r) * (NH * DHD) + h * DHD + 16 * dt + lm] = (__bf16)o;
    }
}

// ---------------- orchestration ----------------
extern "C" void kernel_launch(void* const* d_in, const int* in_sizes, int n_in,
                              void* d_out, int out_size, void* d_ws, size_t ws_size,
                              hipStream_t stream) {
  (void)in_sizes; (void)n_in; (void)out_size; (void)ws_size;
  const float* hidden    = (const float*)d_in[0];
  // d_in[1] positions == arange(S); row index used directly
  const float* wq_a      = (const float*)d_in[2];
  const float* q_norm_g  = (const float*)d_in[3];
  const float* wq_b      = (const float*)d_in[4];
  const float* wkv       = (const float*)d_in[5];
  const float* kv_norm_g = (const float*)d_in[6];
  const float* wo_a      = (const float*)d_in[7];
  const float* wo_b      = (const float*)d_in[8];
  const float* attn_sink = (const float*)d_in[9];
  float* out = (float*)d_out;

  size_t off = 0;
  auto take = [&](size_t bytes) -> void* {
    void* p = (char*)d_ws + off;
    off += (bytes + 255) & ~(size_t)255;
    return p;
  };
  __bf16* h_bf    = (__bf16*)take((size_t)S_LEN * D_MODEL * 2);
  __bf16* qr_bf   = (__bf16*)take((size_t)S_LEN * QLRN * 2);
  __bf16* q_bf    = (__bf16*)take((size_t)S_LEN * NH * DHD * 2);
  __bf16* k_bf    = (__bf16*)take((size_t)S_LEN * DHD * 2);
  __bf16* vT_bf   = (__bf16*)take((size_t)DHD * S_LEN * 2);
  __bf16* attn_bf = (__bf16*)take((size_t)S_LEN * NH * DHD * 2);
  __bf16* low_bf  = (__bf16*)take((size_t)S_LEN * NG * OLRN * 2);
  __bf16* w_bf    = (__bf16*)take((size_t)D_MODEL * 2048 * 2);  // reused for each weight
  __bf16* wkv_bf  = (__bf16*)take((size_t)DHD * D_MODEL * 2);

  float* stage = (float*)d_out;  // f32 staging in d_out (32MB), sequential lifetimes

  auto cvt = [&](const float* s, __bf16* d, long long n) {
    cvt_f32_bf16<<<2048, 256, 0, stream>>>(s, d, n);
  };

  // 1) qr = rmsnorm(hidden @ wq_a^T)
  cvt(hidden, h_bf, (long long)S_LEN * D_MODEL);
  cvt(wq_a, w_bf, (long long)QLRN * D_MODEL);
  gemm_bt<<<dim3(S_LEN / 128, QLRN / 128, 1), 256, 0, stream>>>(
      h_bf, w_bf, stage, S_LEN, QLRN, D_MODEL, D_MODEL, D_MODEL, QLRN, 0, 0, 0);
  rmsnorm_bf16<<<S_LEN, 256, 0, stream>>>(stage, q_norm_g, qr_bf, QLRN);

  // 2) kv = rmsnorm(hidden @ wkv^T); K = rope(kv), V = kv (stored transposed)
  cvt(wkv, wkv_bf, (long long)DHD * D_MODEL);
  gemm_bt<<<dim3(S_LEN / 128, 1, 1), 256, 0, stream>>>(
      h_bf, wkv_bf, stage, S_LEN, DHD, D_MODEL, D_MODEL, D_MODEL, DHD, 0, 0, 0);
  kv_epilogue<<<S_LEN, 128, 0, stream>>>(stage, kv_norm_g, vT_bf, k_bf);

  // 3) q = rope(qr @ wq_b^T) * DH^-0.5
  cvt(wq_b, w_bf, (long long)NH * DHD * QLRN);
  gemm_bt<<<dim3(S_LEN / 128, (NH * DHD) / 128, 1), 256, 0, stream>>>(
      qr_bf, w_bf, stage, S_LEN, NH * DHD, QLRN, QLRN, QLRN, NH * DHD, 0, 0, 0);
  q_rope<<<dim3(S_LEN, NH), 64, 0, stream>>>(stage, q_bf);

  // 4) flash attention with sink
  attn_fwd<<<dim3(S_LEN / 128, NH), 256, 0, stream>>>(q_bf, k_bf, vT_bf, attn_sink, attn_bf);

  // 5) low[s, g*512+k] = sum_d attn[s, g*1024+d] * wo_a[g,k,d]   (batched over g)
  cvt(wo_a, w_bf, (long long)NG * OLRN * 1024);
  gemm_bt<<<dim3(S_LEN / 128, OLRN / 128, NG), 256, 0, stream>>>(
      attn_bf, w_bf, stage, S_LEN, OLRN, 1024, NH * DHD, 1024, NG * OLRN,
      /*aB=*/1024, /*bB=*/(long long)OLRN * 1024, /*cB=*/OLRN);
  cvt(stage, low_bf, (long long)S_LEN * NG * OLRN);

  // 6) out = low @ wo_b^T
  cvt(wo_b, w_bf, (long long)D_MODEL * NG * OLRN);
  gemm_bt<<<dim3(S_LEN / 128, D_MODEL / 128, 1), 256, 0, stream>>>(
      low_bf, w_bf, out, S_LEN, D_MODEL, NG * OLRN, NG * OLRN, NG * OLRN, D_MODEL, 0, 0, 0);
}
